// SwinTransformerBlockImpl_67156108640567
// MI455X (gfx1250) — matrix operand order repaired and checked by reading, not compiled, not run on hardware
//
#include <hip/hip_runtime.h>
#include <hip/hip_bf16.h>
#include <math.h>

typedef __bf16 bf16_t;
typedef __attribute__((ext_vector_type(16))) __bf16 v16bf;
typedef __attribute__((ext_vector_type(8)))  float  v8f;

#define WS7    7
#define SHIFT3 3
#define HEADS  3
#define CDIM   96
#define HID    384
#define NTOK   49
#define NPAD   64
#define HD     32
#define IMG    112
#define BATCH  16
#define LTOK   12544      // 112*112
#define TOKALL 200704     // 16*12544

// ---------------- wave helpers (wave32) ----------------
__device__ __forceinline__ float wred_sum(float v) {
#pragma unroll
  for (int o = 16; o > 0; o >>= 1) v += __shfl_xor(v, o, 32);
  return v;
}
__device__ __forceinline__ float wred_max(float v) {
#pragma unroll
  for (int o = 16; o > 0; o >>= 1) v = fmaxf(v, __shfl_xor(v, o, 32));
  return v;
}

__device__ __forceinline__ v8f wmma_bf16(v16bf a, v16bf b, v8f c) {
  // D = A(16x32) * B(32x16) + C, f32 accumulate
  return __builtin_amdgcn_wmma_f32_16x16x32_bf16(false, a, false, b, (short)0, c,
                                                 false, false);
}

// ---------------- WMMA fragment loaders (CDNA5 layouts) ----------------
// A fragment 16x32 (MxK). base = tile origin, row-major, stride ld.
// lane l: M = l&15 ; lanes 0-15 hold K 0-7 / 16-23, lanes 16-31 hold K 8-15 / 24-31.
// Each lane reads two contiguous 16B runs -> merges to ds_load_b128 / global_load_b128.
__device__ __forceinline__ v16bf load_a(const bf16_t* base, int ld, int lane) {
  const int m  = lane & 15;
  const int hi = (lane >> 4) << 3;
  const bf16_t* p = base + m * ld + hi;
  v16bf a;
#pragma unroll
  for (int e = 0; e < 16; ++e) a[e] = p[((e >> 3) << 4) + (e & 7)];
  return a;
}
// B fragment 32x16 (KxN) where B[k][n] = W[n0+n][k0+k] (W row-major NxK, stride ld).
// lane l: N = l&15 ; lanes 0-15 hold K 0-7 / 16-23, lanes 16-31 hold K 8-15 / 24-31.
// Each lane reads two contiguous 16B runs -> two b128 loads.
__device__ __forceinline__ v16bf load_bT(const bf16_t* base, int ld, int lane) {
  const int n  = lane & 15;
  const int kb = (lane >> 4) << 3;
  const bf16_t* p = base + n * ld + kb;
  v16bf b;
#pragma unroll
  for (int e = 0; e < 16; ++e) b[e] = p[((e >> 3) << 4) + (e & 7)];
  return b;
}

// prefetch a byte range into cache (global_prefetch_b8), one cacheline per 256B step
__device__ __forceinline__ void prefetch_range(const void* base, int bytes, int tid, int nthr) {
  const char* p = (const char*)base;
  for (int o = tid * 256; o < bytes; o += nthr * 256)
    __builtin_prefetch(p + o, 0, 0);
}

// ---------------- kernel 0: weight cast f32 -> bf16 ----------------
__global__ void cvt_f32_bf16(const float* __restrict__ s, bf16_t* __restrict__ d, int n) {
  int i = blockIdx.x * 256 + threadIdx.x;
  if (i < n) d[i] = (bf16_t)s[i];
}

// ---------------- kernel 1: LN1 + shifted-window attention + proj + residual ----------------
__global__ __launch_bounds__(128) void swin_attn_kernel(
    const float* __restrict__ x, const float* __restrict__ mask,
    const float* __restrict__ n1g, const float* __restrict__ n1b,
    const bf16_t* __restrict__ qkvW, const float* __restrict__ qkvB,
    const float* __restrict__ relTab, const bf16_t* __restrict__ projW,
    const float* __restrict__ projB, float* __restrict__ xmid)
{
  __shared__ bf16_t sXn[NPAD * CDIM];   // 12288 B normalized shifted window
  __shared__ bf16_t sQ[NPAD * HD];      //  4096 B per-head Q (pre-scaled)
  __shared__ bf16_t sK[NPAD * HD];      //  4096 B
  __shared__ bf16_t sVt[HD * NPAD];     //  4096 B per-head V TRANSPOSED (ch x tok)
  __shared__ float  sS[NPAD * NPAD];    // 16384 B attention logits
  __shared__ bf16_t sP[NPAD * NPAD];    //  8192 B softmax probs
  __shared__ bf16_t sO[NPAD * CDIM];    // 12288 B attention output (all heads)

  const int wg   = blockIdx.x;        // global window id [0, 4096)
  const int b    = wg >> 8;           // image
  const int w    = wg & 255;          // window within image (16x16)
  const int bh   = w >> 4;
  const int bw   = w & 15;
  const int lane = threadIdx.x & 31;
  const int wave = threadIdx.x >> 5;

  // warm the weight working set while LN runs (global_prefetch_b8)
  prefetch_range(qkvW, 3 * CDIM * CDIM * 2, threadIdx.x, 128);
  prefetch_range(projW, CDIM * CDIM * 2, threadIdx.x, 128);

  // ---- LN1 + cyclic-shift gather -> sXn (rows 49..63 zero-padded) ----
  for (int t = wave; t < NPAD; t += 4) {
    if (t < NTOK) {
      const int r = t / WS7, c = t % WS7;
      const int sh = bh * WS7 + r, sw = bw * WS7 + c;
      const int oh = (sh + SHIFT3) % IMG;
      const int ow = (sw + SHIFT3) % IMG;
      const float* row = x + ((size_t)b * LTOK + (size_t)oh * IMG + ow) * CDIM;
      float v0 = row[lane], v1 = row[lane + 32], v2 = row[lane + 64];
      float mean = wred_sum(v0 + v1 + v2) * (1.0f / CDIM);
      float d0 = v0 - mean, d1 = v1 - mean, d2 = v2 - mean;
      float var = wred_sum(d0 * d0 + d1 * d1 + d2 * d2) * (1.0f / CDIM);
      float inv = rsqrtf(var + 1e-5f);
      sXn[t * CDIM + lane]      = (bf16_t)(d0 * inv * n1g[lane]      + n1b[lane]);
      sXn[t * CDIM + lane + 32] = (bf16_t)(d1 * inv * n1g[lane + 32] + n1b[lane + 32]);
      sXn[t * CDIM + lane + 64] = (bf16_t)(d2 * inv * n1g[lane + 64] + n1b[lane + 64]);
    } else {
      sXn[t * CDIM + lane]      = (bf16_t)0.0f;
      sXn[t * CDIM + lane + 32] = (bf16_t)0.0f;
      sXn[t * CDIM + lane + 64] = (bf16_t)0.0f;
    }
  }
  __syncthreads();

  const float qscale = 0.17677669529663687f;  // 1/sqrt(hd=32)

  for (int h = 0; h < HEADS; ++h) {
    // ---- QKV for this head: X(64x96) @ W^T, cols = which*96 + h*32 + [0,32) ----
    for (int j = wave; j < 24; j += 4) {       // wave-uniform tile jobs
      const int rt    = j & 3;                 // row tile
      const int ct    = j >> 2;                // 0..5
      const int which = ct >> 1;               // 0=Q 1=K 2=V
      const int halfc = ct & 1;
      const int gcol  = which * CDIM + h * HD + halfc * 16;
      v8f acc = {};
#pragma unroll
      for (int kk = 0; kk < 3; ++kk) {
        v16bf a  = load_a(sXn + rt * 16 * CDIM + kk * 32, CDIM, lane);
        v16bf bb = load_bT(qkvW + (size_t)gcol * CDIM + kk * 32, CDIM, lane);
        acc = wmma_bf16(a, bb, acc);
      }
      const int n  = lane & 15;
      const int mb = (lane >> 4) << 3;
      const float bias = qkvB[gcol + n];
      if (which == 2) {
        // V stored transposed (channel-major) so the P@V B-fragments are contiguous
#pragma unroll
        for (int r = 0; r < 8; ++r)
          sVt[(halfc * 16 + n) * NPAD + rt * 16 + mb + r] = (bf16_t)(acc[r] + bias);
      } else {
        bf16_t* dst = (which == 0) ? sQ : sK;
        const float scl = (which == 0) ? qscale : 1.0f;
#pragma unroll
        for (int r = 0; r < 8; ++r)
          dst[(rt * 16 + mb + r) * HD + halfc * 16 + n] = (bf16_t)((acc[r] + bias) * scl);
      }
    }
    __syncthreads();

    // ---- S = Q @ K^T + rel_bias + shift_mask (K=32 -> one WMMA per tile) ----
    for (int j = wave; j < 16; j += 4) {
      const int rt = j >> 2, ctl = j & 3;
      v16bf a  = load_a(sQ + rt * 16 * HD, HD, lane);
      v16bf bb = load_bT(sK + ctl * 16 * HD, HD, lane);
      v8f acc = {};
      acc = wmma_bf16(a, bb, acc);
      const int n  = lane & 15;
      const int mb = (lane >> 4) << 3;
#pragma unroll
      for (int r = 0; r < 8; ++r) {
        const int i  = rt * 16 + mb + r;
        const int jj = ctl * 16 + n;
        float val;
        if (i < NTOK && jj < NTOK) {
          const int ri = i / WS7, ci = i % WS7;
          const int rj = jj / WS7, cj = jj % WS7;
          const int idx = (ri - rj + WS7 - 1) * (2 * WS7 - 1) + (ci - cj + WS7 - 1);
          val = acc[r] + relTab[idx * HEADS + h]
                       + mask[((size_t)w * NTOK + i) * NTOK + jj];
        } else {
          val = -1e30f;   // kill padded key columns
        }
        sS[i * NPAD + jj] = val;
      }
    }
    __syncthreads();

    // ---- row softmax (one wave per row, 2 cols per lane) ----
    for (int rrow = wave; rrow < NPAD; rrow += 4) {
      float a0 = sS[rrow * NPAD + lane];
      float a1 = sS[rrow * NPAD + lane + 32];
      float mx = wred_max(fmaxf(a0, a1));
      float e0 = __expf(a0 - mx), e1 = __expf(a1 - mx);
      float inv = 1.0f / wred_sum(e0 + e1);
      sP[rrow * NPAD + lane]      = (bf16_t)(e0 * inv);
      sP[rrow * NPAD + lane + 32] = (bf16_t)(e1 * inv);
    }
    __syncthreads();

    // ---- O_h = P(64x64) @ V(64x32), K=64 -> 2 WMMA steps; B from V^T is contiguous ----
    for (int j = wave; j < 8; j += 4) {
      const int rt = j >> 1, ctl = j & 1;
      v8f acc = {};
#pragma unroll
      for (int kk = 0; kk < 2; ++kk) {
        v16bf a  = load_a(sP + rt * 16 * NPAD + kk * 32, NPAD, lane);
        // B[k][n] = V[k][n] = Vt[n][k] -> load_bT over Vt (ld = NPAD)
        v16bf bb = load_bT(sVt + (size_t)(ctl * 16) * NPAD + kk * 32, NPAD, lane);
        acc = wmma_bf16(a, bb, acc);
      }
      const int n  = lane & 15;
      const int mb = (lane >> 4) << 3;
#pragma unroll
      for (int r = 0; r < 8; ++r)
        sO[(rt * 16 + mb + r) * CDIM + h * HD + ctl * 16 + n] = (bf16_t)acc[r];
    }
    __syncthreads();
  }

  // ---- proj + residual + reverse-shift scatter -> xmid ----
  for (int j = wave; j < 24; j += 4) {
    const int rt = j & 3;
    const int ct = j >> 2;                 // 0..5
    v8f acc = {};
#pragma unroll
    for (int kk = 0; kk < 3; ++kk) {
      v16bf a  = load_a(sO + rt * 16 * CDIM + kk * 32, CDIM, lane);
      v16bf bb = load_bT(projW + (size_t)(ct * 16) * CDIM + kk * 32, CDIM, lane);
      acc = wmma_bf16(a, bb, acc);
    }
    const int n  = lane & 15;
    const int mb = (lane >> 4) << 3;
    const float pb = projB[ct * 16 + n];
#pragma unroll
    for (int r = 0; r < 8; ++r) {
      const int i = rt * 16 + mb + r;
      if (i < NTOK) {
        const int rr = i / WS7, cc = i % WS7;
        const int sh = bh * WS7 + rr, sw = bw * WS7 + cc;
        const int oh = (sh + SHIFT3) % IMG;
        const int ow = (sw + SHIFT3) % IMG;
        const size_t off = ((size_t)b * LTOK + (size_t)oh * IMG + ow) * CDIM + ct * 16 + n;
        xmid[off] = x[off] + acc[r] + pb;   // shortcut + attn_out (reverse shift == same loc)
      }
    }
  }
}

// ---------------- kernel 2: LN2 + MLP (FC1 -> GELU -> FC2) + residual ----------------
__global__ __launch_bounds__(128) void swin_mlp_kernel(
    const float* __restrict__ xmid,
    const float* __restrict__ n2g, const float* __restrict__ n2b,
    const bf16_t* __restrict__ fc1W, const float* __restrict__ fc1B,
    const bf16_t* __restrict__ fc2W, const float* __restrict__ fc2B,
    float* __restrict__ out)
{
  __shared__ bf16_t sXn[64 * CDIM];   // 12288 B
  __shared__ bf16_t sH1[64 * HID];    // 49152 B

  const size_t row0 = (size_t)blockIdx.x * 64;
  const int lane = threadIdx.x & 31;
  const int wave = threadIdx.x >> 5;

  // warm the FC weights while LN runs
  prefetch_range(fc1W, HID * CDIM * 2, threadIdx.x, 128);
  prefetch_range(fc2W, CDIM * HID * 2, threadIdx.x, 128);

  // ---- LN2 -> bf16 ----
  for (int t = wave; t < 64; t += 4) {
    const float* row = xmid + (row0 + t) * CDIM;
    float v0 = row[lane], v1 = row[lane + 32], v2 = row[lane + 64];
    float mean = wred_sum(v0 + v1 + v2) * (1.0f / CDIM);
    float d0 = v0 - mean, d1 = v1 - mean, d2 = v2 - mean;
    float var = wred_sum(d0 * d0 + d1 * d1 + d2 * d2) * (1.0f / CDIM);
    float inv = rsqrtf(var + 1e-5f);
    sXn[t * CDIM + lane]      = (bf16_t)(d0 * inv * n2g[lane]      + n2b[lane]);
    sXn[t * CDIM + lane + 32] = (bf16_t)(d1 * inv * n2g[lane + 32] + n2b[lane + 32]);
    sXn[t * CDIM + lane + 64] = (bf16_t)(d2 * inv * n2g[lane + 64] + n2b[lane + 64]);
  }
  __syncthreads();

  // ---- FC1 + exact GELU ----
  for (int j = wave; j < 96; j += 4) {
    const int rt = j & 3;
    const int ct = j >> 2;              // 0..23
    v8f acc = {};
#pragma unroll
    for (int kk = 0; kk < 3; ++kk) {
      v16bf a  = load_a(sXn + rt * 16 * CDIM + kk * 32, CDIM, lane);
      v16bf bb = load_bT(fc1W + (size_t)(ct * 16) * CDIM + kk * 32, CDIM, lane);
      acc = wmma_bf16(a, bb, acc);
    }
    const int n  = lane & 15;
    const int mb = (lane >> 4) << 3;
    const float bias = fc1B[ct * 16 + n];
#pragma unroll
    for (int r = 0; r < 8; ++r) {
      float v = acc[r] + bias;
      v = 0.5f * v * (1.0f + erff(v * 0.70710678118654752f));
      sH1[(rt * 16 + mb + r) * HID + ct * 16 + n] = (bf16_t)v;
    }
  }
  __syncthreads();

  // ---- FC2 + residual -> out ----
  for (int j = wave; j < 24; j += 4) {
    const int rt = j & 3;
    const int ct = j >> 2;              // 0..5
    v8f acc = {};
#pragma unroll
    for (int kk = 0; kk < 12; ++kk) {
      v16bf a  = load_a(sH1 + rt * 16 * HID + kk * 32, HID, lane);
      v16bf bb = load_bT(fc2W + (size_t)(ct * 16) * HID + kk * 32, HID, lane);
      acc = wmma_bf16(a, bb, acc);
    }
    const int n  = lane & 15;
    const int mb = (lane >> 4) << 3;
    const float bias = fc2B[ct * 16 + n];
#pragma unroll
    for (int r = 0; r < 8; ++r) {
      const size_t idx = (row0 + rt * 16 + mb + r) * CDIM + ct * 16 + n;
      out[idx] = xmid[idx] + acc[r] + bias;
    }
  }
}

// ---------------- host launcher ----------------
extern "C" void kernel_launch(void* const* d_in, const int* in_sizes, int n_in,
                              void* d_out, int out_size, void* d_ws, size_t ws_size,
                              hipStream_t stream) {
  (void)in_sizes; (void)n_in; (void)out_size; (void)ws_size;
  const float* x     = (const float*)d_in[0];
  const float* mask  = (const float*)d_in[1];
  const float* n1g   = (const float*)d_in[2];
  const float* n1b   = (const float*)d_in[3];
  const float* qkvw  = (const float*)d_in[4];
  const float* qkvb  = (const float*)d_in[5];
  const float* rtab  = (const float*)d_in[6];
  const float* projw = (const float*)d_in[7];
  const float* projb = (const float*)d_in[8];
  const float* n2g   = (const float*)d_in[9];
  const float* n2b   = (const float*)d_in[10];
  const float* fc1w  = (const float*)d_in[11];
  const float* fc1b  = (const float*)d_in[12];
  const float* fc2w  = (const float*)d_in[13];
  const float* fc2b  = (const float*)d_in[14];

  // workspace layout (all 256B aligned)
  char*   ws    = (char*)d_ws;
  float*  xmid  = (float*)ws;                 // 200704*96*4 = 77,070,336 B
  size_t  off   = (size_t)TOKALL * CDIM * sizeof(float);
  bf16_t* qkvWb = (bf16_t*)(ws + off);  off += (size_t)(3 * CDIM) * CDIM * 2;   // 55,296
  bf16_t* projWb= (bf16_t*)(ws + off);  off += (size_t)CDIM * CDIM * 2;         // 18,432
  bf16_t* fc1Wb = (bf16_t*)(ws + off);  off += (size_t)HID * CDIM * 2;          // 73,728
  bf16_t* fc2Wb = (bf16_t*)(ws + off);  off += (size_t)CDIM * HID * 2;          // 73,728

  // kernel 0: one-shot weight casts (cheap; weights stay L2-resident)
  cvt_f32_bf16<<<(3 * CDIM * CDIM + 255) / 256, 256, 0, stream>>>(qkvw, qkvWb, 3 * CDIM * CDIM);
  cvt_f32_bf16<<<(CDIM * CDIM + 255) / 256, 256, 0, stream>>>(projw, projWb, CDIM * CDIM);
  cvt_f32_bf16<<<(HID * CDIM + 255) / 256, 256, 0, stream>>>(fc1w, fc1Wb, HID * CDIM);
  cvt_f32_bf16<<<(CDIM * HID + 255) / 256, 256, 0, stream>>>(fc2w, fc2Wb, CDIM * HID);

  // kernel 1: one block per shifted window (16 images * 256 windows)
  swin_attn_kernel<<<BATCH * 256, 128, 0, stream>>>(
      x, mask, n1g, n1b, qkvWb, qkvb, rtab, projWb, projb, xmid);

  // kernel 2: 64-token row blocks over all 200704 tokens
  swin_mlp_kernel<<<TOKALL / 64, 128, 0, stream>>>(
      xmid, n2g, n2b, fc1Wb, fc1b, fc2Wb, fc2b, (float*)d_out);
}
